// dmap_46316927320313
// MI455X (gfx1250) — matrix operand order repaired and checked by reading, not compiled, not run on hardware
//
#include <hip/hip_runtime.h>

// Problem constants (from reference)
#define BB   2048
#define DD   64
#define NN   16384
#define HH   4
#define DOUTC 256
#define EPSC 1e-12f

#define KC  256          // N-chunk per iteration
#define BT  16           // batch rows per workgroup
#define EST 260          // eBuf row stride in floats (pad: 260 mod 64 = 4 -> conflict-free row gathers)

typedef __bf16 v16bf __attribute__((ext_vector_type(16)));
typedef float  v8f   __attribute__((ext_vector_type(8)));

// workspace layout (bytes)
#define WS_R2    0L
#define WS_INVQ  (WS_R2   + (long)NN*4)          // H*N f32
#define WS_RHI   (WS_INVQ + (long)HH*NN*4)       // N*D bf16 packed (B-layout)
#define WS_RLO   (WS_RHI  + (long)NN*DD*2)
#define WS_WPK   (WS_RLO  + (long)NN*DD*2)       // H*N*DOUT bf16 packed (B-layout)

__device__ __forceinline__ v8f vzero8() {
    v8f z;
    #pragma unroll
    for (int i = 0; i < 8; ++i) z[i] = 0.f;
    return z;
}

// ---------------- prep kernels ----------------

__global__ void k_prep_invq(const float* __restrict__ q, float* __restrict__ invq) {
    int i = blockIdx.x * 256 + threadIdx.x;
    if (i < HH * NN) {
        float v = fmaxf(q[i], 0.f) + EPSC;   // alpha == 1 -> q^-alpha = 1/q
        invq[i] = 1.f / v;
    }
}

__global__ void k_prep_r2(const float* __restrict__ R, float* __restrict__ r2) {
    int n = blockIdx.x * 256 + threadIdx.x;
    if (n < NN) {
        float s = 0.f;
        for (int d = 0; d < DD; ++d) { float r = R[n * DD + d]; s += r * r; }
        r2[n] = s;
    }
}

// Pack R^T (D x N) into bf16 B-matrix fragments, split hi/lo for accuracy.
// Layout: [nTile(N/16)][kt(2)][lane(32)][e(16)] ; lane holds col n = nTile*16+(lane&15),
// element e holds K = kt*32 + e + (lane&16 ? 16 : 0).
__global__ void k_pack_R(const float* __restrict__ R,
                         unsigned short* __restrict__ hi_us,
                         unsigned short* __restrict__ lo_us) {
    __bf16* hi = (__bf16*)hi_us;  __bf16* lo = (__bf16*)lo_us;
    int g = blockIdx.x * 256 + threadIdx.x;
    int lane = g & 31, kt = (g >> 5) & 1, nt = g >> 6;
    if (nt >= NN / 16) return;
    int col = nt * 16 + (lane & 15);
    int kb  = kt * 32 + ((lane & 16) ? 16 : 0);
    long base = (long)g * 16;
    #pragma unroll
    for (int e = 0; e < 16; ++e) {
        float r = R[(long)col * DD + kb + e];
        __bf16 h = (__bf16)r;
        hi[base + e] = h;
        lo[base + e] = (__bf16)(r - (float)h);
    }
}

// Pack W (H, N, DOUT) into bf16 B-matrix fragments.
// Layout: [h][ktG(N/32)][ct(DOUT/16)][lane(32)][e(16)]
__global__ void k_pack_W(const float* __restrict__ W, unsigned short* __restrict__ wpk_us) {
    __bf16* wpk = (__bf16*)wpk_us;
    int g = blockIdx.x * 256 + threadIdx.x;           // ((h*512+ktG)*16+ct)*32+lane
    int lane = g & 31, ct = (g >> 5) & 15, ktG = (g >> 9) & 511, h = g >> 18;
    if (h >= HH) return;
    int n    = ktG * 32 + ((lane & 16) ? 16 : 0);
    int dcol = ct * 16 + (lane & 15);
    const float* Wh = W + (long)h * NN * DOUTC;
    long base = (long)g * 16;
    #pragma unroll
    for (int e = 0; e < 16; ++e)
        wpk[base + e] = (__bf16)Wh[(long)(n + e) * DOUTC + dcol];
}

// ---------------- fused main kernel ----------------
// grid = B/BT workgroups of 256 threads (8 waves).
// wave -> (head = wave>>1, dout half = wave&1); each wave owns a 16x128 f32 accumulator.

__global__ __launch_bounds__(256, 1) void k_fused(
    const float* __restrict__ x,  const float* __restrict__ r2g,
    const float* __restrict__ invqg,
    const unsigned short* __restrict__ rhig_us,
    const unsigned short* __restrict__ rlog_us,
    const unsigned short* __restrict__ wpk_us,
    float* __restrict__ out)
{
    const __bf16* rhig = (const __bf16*)rhig_us;
    const __bf16* rlog = (const __bf16*)rlog_us;
    const __bf16* wpk  = (const __bf16*)wpk_us;

    __shared__ float xs[BT][DD];
    __shared__ float xs2[BT];
    __shared__ float eBuf[BT][EST];
    __shared__ float invqCh[HH][KC];
    __shared__ float r2Ch[KC];
    __shared__ float part[BT][16];
    __shared__ float mrun[BT], scaleS[BT];
    __shared__ float wsum[BT][HH], invs[BT][HH];

    const int t = threadIdx.x, lane = t & 31, wave = t >> 5;
    const int head = wave >> 1, dhalf = wave & 1;
    const int rlo = lane & 15, khi = (lane & 16) ? 8 : 0;
    const int b0 = blockIdx.x * BT;

    for (int i = t; i < BT * DD; i += 256) {
        int r = i / DD, d = i % DD;
        xs[r][d] = x[(long)(b0 + r) * DD + d];
    }
    if (t < BT) mrun[t] = -1e30f;
    if (t < BT * HH) (&wsum[0][0])[t] = 0.f;
    __syncthreads();
    if (t < BT) {
        float s = 0.f;
        for (int d = 0; d < DD; ++d) { float v = xs[t][d]; s += v * v; }
        xs2[t] = s;
    }

    // x A-fragments (constant over all chunks): 16x32 bf16 tiles, hi/lo split.
    v16bf xhi[2], xlo[2];
    #pragma unroll
    for (int kt = 0; kt < 2; ++kt) {
        #pragma unroll
        for (int e = 0; e < 16; ++e) {
            int k = kt * 32 + e + 2 * khi;   // A-layout K striping
            float v = xs[rlo][k];
            __bf16 h = (__bf16)v;
            xhi[kt][e] = h;
            xlo[kt][e] = (__bf16)(v - (float)h);
        }
    }

    v8f acc[8];
    #pragma unroll
    for (int ct = 0; ct < 8; ++ct) acc[ct] = vzero8();

    for (int ch = 0; ch < NN / KC; ++ch) {
        const int n0 = ch * KC;
        // stage per-chunk tables (t == 0..255 == KC)
        r2Ch[t] = r2g[n0 + t];
        #pragma unroll
        for (int h = 0; h < HH; ++h) invqCh[h][t] = invqg[(long)h * NN + n0 + t];
        __syncthreads();   // also fences prev stage-2 reads of eBuf

        // ---- stage 1: cross = x @ R^T via split-bf16 WMMA; write logits to LDS
        #pragma unroll
        for (int sub = 0; sub < 2; ++sub) {
            int ntl = wave * 2 + sub;          // 16-col tile within chunk
            int ntG = ch * 16 + ntl;
            v8f c = vzero8();
            #pragma unroll
            for (int kt = 0; kt < 2; ++kt) {
                const v16bf bh = *(const v16bf*)(rhig + ((long)(ntG * 2 + kt) * 32 + lane) * 16);
                const v16bf bl = *(const v16bf*)(rlog + ((long)(ntG * 2 + kt) * 32 + lane) * 16);
                c = __builtin_amdgcn_wmma_f32_16x16x32_bf16(false, xhi[kt], false, bh, (short)0, c, false, false);
                c = __builtin_amdgcn_wmma_f32_16x16x32_bf16(false, xlo[kt], false, bh, (short)0, c, false, false);
                c = __builtin_amdgcn_wmma_f32_16x16x32_bf16(false, xhi[kt], false, bl, (short)0, c, false, false);
            }
            float R2c = r2Ch[ntl * 16 + rlo];
            #pragma unroll
            for (int v = 0; v < 8; ++v) {
                int row = v + khi;
                float l = -fmaxf(xs2[row] - 2.f * c[v] + R2c, 0.f);
                eBuf[row][ntl * 16 + rlo] = l;
            }
        }
        __syncthreads();

        // ---- pass 2: per-row chunk max, online-softmax rescale factors
        { int r = t >> 4, cg = t & 15; float lm = -1e30f;
          for (int j = 0; j < 16; ++j) lm = fmaxf(lm, eBuf[r][cg * 16 + j]);
          part[r][cg] = lm; }
        __syncthreads();
        if (t < BT) {
            float cm = -1e30f;
            for (int j = 0; j < 16; ++j) cm = fmaxf(cm, part[t][j]);
            float mo = mrun[t], mn = fmaxf(mo, cm);
            float sc = __expf(mo - mn);
            mrun[t] = mn; scaleS[t] = sc;
            for (int h = 0; h < HH; ++h) wsum[t][h] *= sc;
        }
        __syncthreads();

        // ---- pass 3: e = exp(l - m), per-head weighted row sums
        { int r = t >> 4, cg = t & 15; float mn = mrun[r];
          float s[HH] = {0.f, 0.f, 0.f, 0.f};
          for (int j = 0; j < 16; ++j) {
              int idx = cg * 16 + j;
              float e = __expf(eBuf[r][idx] - mn);
              eBuf[r][idx] = e;
              #pragma unroll
              for (int h = 0; h < HH; ++h) s[h] += e * invqCh[h][idx];
          }
          #pragma unroll
          for (int h = 0; h < HH; ++h) atomicAdd(&wsum[r][h], s[h]); }
        __syncthreads();

        // ---- stage 2: acc = acc*scale + (e .* invq_h) @ W_h  (bf16 WMMA, f32 accum)
        {
            float scv[8];
            #pragma unroll
            for (int v = 0; v < 8; ++v) scv[v] = scaleS[v + khi];
            #pragma unroll
            for (int ct = 0; ct < 8; ++ct) {
                #pragma unroll
                for (int v = 0; v < 8; ++v) acc[ct][v] *= scv[v];
            }

            for (int kt = 0; kt < 8; ++kt) {
                v16bf af;
                #pragma unroll
                for (int e = 0; e < 16; ++e) {
                    int k = kt * 32 + e + 2 * khi;
                    af[e] = (__bf16)(eBuf[rlo][k] * invqCh[head][k]);
                }
                const __bf16* wb = wpk +
                    ((((long)head * 512 + (ch * 8 + kt)) * 16 + dhalf * 8) * 32 + lane) * 16;
                #pragma unroll
                for (int ct = 0; ct < 8; ++ct) {
                    const v16bf bf = *(const v16bf*)(wb + (long)ct * 32 * 16);
                    acc[ct] = __builtin_amdgcn_wmma_f32_16x16x32_bf16(false, af, false, bf, (short)0, acc[ct], false, false);
                }
            }
        }
        __syncthreads();
    }

    // ---- finalize: y = acc / (sum_w + eps)
    if (t < BT * HH) { int r = t >> 2, h = t & 3; invs[r][h] = 1.f / (wsum[r][h] + EPSC); }
    __syncthreads();
    {
        float fv[8];
        #pragma unroll
        for (int v = 0; v < 8; ++v) fv[v] = invs[v + khi][head];
        #pragma unroll
        for (int ct = 0; ct < 8; ++ct) {
            int dc = dhalf * 128 + ct * 16 + rlo;
            #pragma unroll
            for (int v = 0; v < 8; ++v) {
                int row = v + khi;
                out[(((long)(b0 + row)) * HH + head) * DOUTC + dc] = acc[ct][v] * fv[v];
            }
        }
    }
}

// ---------------- launch ----------------

extern "C" void kernel_launch(void* const* d_in, const int* in_sizes, int n_in,
                              void* d_out, int out_size, void* d_ws, size_t ws_size,
                              hipStream_t stream) {
    const float* x = (const float*)d_in[0];
    const float* R = (const float*)d_in[1];
    const float* q = (const float*)d_in[2];
    const float* W = (const float*)d_in[3];
    float* out = (float*)d_out;
    char* ws = (char*)d_ws;

    float* r2   = (float*)(ws + WS_R2);
    float* invq = (float*)(ws + WS_INVQ);
    unsigned short* rhi = (unsigned short*)(ws + WS_RHI);
    unsigned short* rlo = (unsigned short*)(ws + WS_RLO);
    unsigned short* wpk = (unsigned short*)(ws + WS_WPK);

    k_prep_invq<<<(HH * NN + 255) / 256, 256, 0, stream>>>(q, invq);
    k_prep_r2  <<<(NN + 255) / 256, 256, 0, stream>>>(R, r2);
    k_pack_R   <<<(NN / 16) * 2 * 32 / 256, 256, 0, stream>>>(R, rhi, rlo);
    k_pack_W   <<<HH * 512 * 16 * 32 / 256, 256, 0, stream>>>(W, wpk);
    k_fused    <<<BB / BT, 256, 0, stream>>>(x, r2, invq, rhi, rlo, wpk, out);
}
